// HierarchicalHeteroGraphSage_77403900609229
// MI455X (gfx1250) — matrix operand order repaired and checked by reading, not compiled, not run on hardware
//
#include <hip/hip_runtime.h>
#include <cstdint>
#include <cstddef>

typedef __bf16 bf16;
typedef __attribute__((ext_vector_type(16))) __bf16 v16bf;
typedef __attribute__((ext_vector_type(8)))  __bf16 v8bf;
typedef __attribute__((ext_vector_type(8)))  float  v8f;

#define NP_  100000
#define NA_  50000
#define NE_  500000
#define D_IN 128
#define D_H  256
#define D_OUT 349

// ---------------------------------------------------------------------------
// utility kernels
// ---------------------------------------------------------------------------
__global__ __launch_bounds__(256) void k_zero_f32(float* __restrict__ p, size_t n4) {
  size_t t = (size_t)blockIdx.x * blockDim.x + threadIdx.x;
  if (t < n4) ((float4*)p)[t] = make_float4(0.f, 0.f, 0.f, 0.f);
}

__global__ __launch_bounds__(256) void k_count(const int* __restrict__ dst,
                                               float* __restrict__ cnt, int E) {
  int e = blockIdx.x * blockDim.x + threadIdx.x;
  if (e < E) atomicAdd(&cnt[dst[e]], 1.0f);
}

// scatter-add of D-dim rows: agg[dst[e]] += x[src[e]]
template <int D>
__global__ __launch_bounds__(256) void k_scatter(const float* __restrict__ x,
                                                 const int* __restrict__ src,
                                                 const int* __restrict__ dst,
                                                 float* __restrict__ agg, int E) {
  const int D4 = D / 4;
  long t = (long)blockIdx.x * blockDim.x + threadIdx.x;
  if (t >= (long)E * D4) return;
  int e = (int)(t / D4);
  int c = (int)(t % D4);
  int s = src[e], d = dst[e];
  const float4 v = *(const float4*)(x + (size_t)s * D + c * 4);
  float* p = agg + (size_t)d * D + c * 4;
  atomicAdd(p + 0, v.x);
  atomicAdd(p + 1, v.y);
  atomicAdd(p + 2, v.z);
  atomicAdd(p + 3, v.w);
}

// A_p = [mean_c | mean_w | x_root]  (bf16, row stride 3D)
template <int D>
__global__ __launch_bounds__(256) void k_buildA_paper(
    const float* __restrict__ aggc, const float* __restrict__ cntc,
    const float* __restrict__ aggw, const float* __restrict__ cntw,
    const float* __restrict__ xr, bf16* __restrict__ A, int M) {
  long t = (long)blockIdx.x * blockDim.x + threadIdx.x;
  if (t >= (long)M * D) return;
  int i = (int)(t / D), c = (int)(t % D);
  float ic = 1.0f / fmaxf(cntc[i], 1.0f);
  float iw = 1.0f / fmaxf(cntw[i], 1.0f);
  bf16* Ar = A + (size_t)i * (3 * D);
  Ar[c]         = (bf16)(aggc[(size_t)i * D + c] * ic);
  Ar[D + c]     = (bf16)(aggw[(size_t)i * D + c] * iw);
  Ar[2 * D + c] = (bf16)(xr[(size_t)i * D + c]);
}

// A_a = [mean_r | x_root]  (bf16, row stride 2D)
template <int D>
__global__ __launch_bounds__(256) void k_buildA_author(
    const float* __restrict__ aggr, const float* __restrict__ cntr,
    const float* __restrict__ xr, bf16* __restrict__ A, int M) {
  long t = (long)blockIdx.x * blockDim.x + threadIdx.x;
  if (t >= (long)M * D) return;
  int i = (int)(t / D), c = (int)(t % D);
  float ir = 1.0f / fmaxf(cntr[i], 1.0f);
  bf16* Ar = A + (size_t)i * (2 * D);
  Ar[c]     = (bf16)(aggr[(size_t)i * D + c] * ir);
  Ar[D + c] = (bf16)(xr[(size_t)i * D + c]);
}

__global__ __launch_bounds__(256) void k_cvt_bf16(const float* __restrict__ h,
                                                  bf16* __restrict__ A, long n) {
  long t = (long)blockIdx.x * blockDim.x + threadIdx.x;
  if (t < n) A[t] = (bf16)h[t];
}

// Bt[n][k] over k<3D rows: [Wl1 ; Wl2 ; Wr1+Wr2], weights stored [D][DH] row-major
template <int D>
__global__ __launch_bounds__(256) void k_buildBt3(
    const float* __restrict__ Wl1, const float* __restrict__ Wl2,
    const float* __restrict__ Wr1, const float* __restrict__ Wr2,
    bf16* __restrict__ Bt) {
  int t = blockIdx.x * blockDim.x + threadIdx.x;
  const int K = 3 * D;
  if (t >= D_H * K) return;
  int n = t / K, k = t % K;
  float v;
  if (k < D)            v = Wl1[(size_t)k * D_H + n];
  else if (k < 2 * D)   v = Wl2[(size_t)(k - D) * D_H + n];
  else                  v = Wr1[(size_t)(k - 2 * D) * D_H + n] +
                            Wr2[(size_t)(k - 2 * D) * D_H + n];
  Bt[(size_t)n * K + k] = (bf16)v;
}

template <int D>
__global__ __launch_bounds__(256) void k_buildBt2(
    const float* __restrict__ Wl, const float* __restrict__ Wr,
    bf16* __restrict__ Bt) {
  int t = blockIdx.x * blockDim.x + threadIdx.x;
  const int K = 2 * D;
  if (t >= D_H * K) return;
  int n = t / K, k = t % K;
  float v = (k < D) ? Wl[(size_t)k * D_H + n] : Wr[(size_t)(k - D) * D_H + n];
  Bt[(size_t)n * K + k] = (bf16)v;
}

// W_out [256][349] -> Bt [384][256] (zero-padded rows)
__global__ __launch_bounds__(256) void k_buildBtOut(const float* __restrict__ W,
                                                    bf16* __restrict__ Bt) {
  int t = blockIdx.x * blockDim.x + threadIdx.x;
  if (t >= 384 * D_H) return;
  int n = t / D_H, k = t % D_H;
  float v = (n < D_OUT) ? W[(size_t)k * D_OUT + n] : 0.0f;
  Bt[(size_t)n * D_H + k] = (bf16)v;
}

__global__ __launch_bounds__(256) void k_bias_add(const float* __restrict__ b1,
                                                  const float* __restrict__ b2,
                                                  float* __restrict__ o) {
  int t = blockIdx.x * blockDim.x + threadIdx.x;
  if (t < D_H) o[t] = b1[t] + b2[t];
}

// ---------------------------------------------------------------------------
// bf16 WMMA GEMM:  C[M,N] = relu?( A[M,K](bf16) * Bt[N,K]^T (bf16) + bias[N] )
// block tile 128x128, 8 waves (4m x 2n), wave tile 32x64 = 2x4 WMMA frags
// K must be a multiple of 32; Bt must have >= gridDim.x*128 rows allocated.
// ---------------------------------------------------------------------------
__global__ __launch_bounds__(256) void k_gemm_bf16(
    const bf16* __restrict__ A, const bf16* __restrict__ Bt,
    const float* __restrict__ bias, float* __restrict__ C,
    int M, int N, int K, int relu) {
  // rows padded to 40 bf16 (80B) so all 16B fragment chunks stay aligned
  __shared__ alignas(16) bf16 As[128][40];
  __shared__ alignas(16) bf16 Bs[128][40];

  const int tid  = threadIdx.x;
  const int m0   = blockIdx.y * 128;
  const int n0   = blockIdx.x * 128;
  const int wave = tid >> 5;
  const int lane = tid & 31;
  const int wm   = (wave >> 1) * 32;   // wave tile origin in M (0..96)
  const int wn   = (wave & 1) * 64;    // wave tile origin in N (0/64)
  const int g    = lane >> 4;          // lane group
  const int nl   = lane & 15;

  v8f acc[2][4] = {};

  for (int k0 = 0; k0 < K; k0 += 32) {
    __syncthreads();
    // ---- stage A tile 128x32 (512 chunks of 8 bf16, 2 per thread) ----
#pragma unroll
    for (int i = 0; i < 2; ++i) {
      int chunk = tid + i * 256;
      int r  = chunk >> 2;
      int kc = (chunk & 3) * 8;
      int gr = m0 + r;
      uint4 v = make_uint4(0u, 0u, 0u, 0u);
      if (gr < M) v = *(const uint4*)(A + (size_t)gr * K + k0 + kc);
      *(uint4*)&As[r][kc] = v;
    }
    // ---- stage B tile 128x32 (512 chunks, 2 per thread; rows pre-padded) ----
#pragma unroll
    for (int i = 0; i < 2; ++i) {
      int chunk = tid + i * 256;
      int r  = chunk >> 2;
      int kc = (chunk & 3) * 8;
      *(uint4*)&Bs[r][kc] = *(const uint4*)(Bt + (size_t)(n0 + r) * K + k0 + kc);
    }
    __syncthreads();

    // ---- fragments ----
    v16bf af[2], bff[4];
#pragma unroll
    for (int i = 0; i < 2; ++i) {
      int row = wm + i * 16 + nl;
      // A 16-bit 16x32 layout: group0 -> K {0..7,16..23}, group1 -> K {8..15,24..31}
      v8bf lo = *(const v8bf*)&As[row][g ? 8 : 0];
      v8bf hi = *(const v8bf*)&As[row][g ? 24 : 16];
#pragma unroll
      for (int j = 0; j < 8; ++j) { af[i][j] = lo[j]; af[i][8 + j] = hi[j]; }
    }
#pragma unroll
    for (int i = 0; i < 4; ++i) {
      int row = wn + i * 16 + nl;   // column of B
      // B 16-bit 32x16 layout: group0 -> K {0..7,16..23}, group1 -> K {8..15,24..31}
      v8bf lo = *(const v8bf*)&Bs[row][g ? 8 : 0];
      v8bf hi = *(const v8bf*)&Bs[row][g ? 24 : 16];
#pragma unroll
      for (int j = 0; j < 8; ++j) { bff[i][j] = lo[j]; bff[i][8 + j] = hi[j]; }
    }

#pragma unroll
    for (int i = 0; i < 2; ++i)
#pragma unroll
      for (int j = 0; j < 4; ++j)
        acc[i][j] = __builtin_amdgcn_wmma_f32_16x16x32_bf16(
            false, af[i], false, bff[j], (short)0, acc[i][j], false, false);
  }

  // ---- epilogue: C/D layout — lane nl = column, vgpr r = row (g*8 offset) ----
#pragma unroll
  for (int j = 0; j < 4; ++j) {
    int gn = n0 + wn + j * 16 + nl;
    if (gn >= N) continue;
    float bv = bias ? bias[gn] : 0.0f;
#pragma unroll
    for (int i = 0; i < 2; ++i) {
#pragma unroll
      for (int r = 0; r < 8; ++r) {
        int gm = m0 + wm + i * 16 + g * 8 + r;
        if (gm < M) {
          float v = acc[i][j][r] + bv;
          if (relu) v = fmaxf(v, 0.0f);
          C[(size_t)gm * N + gn] = v;
        }
      }
    }
  }
}

// ---------------------------------------------------------------------------
static inline int cdiv(long a, long b) { return (int)((a + b - 1) / b); }

static void gemm(const bf16* A, const bf16* Bt, const float* bias, float* C,
                 int M, int N, int K, int relu, hipStream_t s) {
  dim3 grid(cdiv(N, 128), cdiv(M, 128));
  k_gemm_bf16<<<grid, 256, 0, s>>>(A, Bt, bias, C, M, N, K, relu);
}

extern "C" void kernel_launch(void* const* d_in, const int* in_sizes, int n_in,
                              void* d_out, int out_size, void* d_ws, size_t ws_size,
                              hipStream_t stream) {
  // ---- inputs (setup_inputs dict order) ----
  const float* x_paper  = (const float*)d_in[0];
  const float* x_author = (const float*)d_in[1];
  const float* l1_Wl_c  = (const float*)d_in[2];
  const float* l1_Wr_c  = (const float*)d_in[3];
  const float* l1_b_c   = (const float*)d_in[4];
  const float* l1_Wl_w  = (const float*)d_in[5];
  const float* l1_Wr_w  = (const float*)d_in[6];
  const float* l1_b_w   = (const float*)d_in[7];
  const float* l1_Wl_r  = (const float*)d_in[8];
  const float* l1_Wr_r  = (const float*)d_in[9];
  const float* l1_b_r   = (const float*)d_in[10];
  const float* l2_Wl_c  = (const float*)d_in[11];
  const float* l2_Wr_c  = (const float*)d_in[12];
  const float* l2_b_c   = (const float*)d_in[13];
  const float* l2_Wl_w  = (const float*)d_in[14];
  const float* l2_Wr_w  = (const float*)d_in[15];
  const float* l2_b_w   = (const float*)d_in[16];
  const float* l2_Wl_r  = (const float*)d_in[17];
  const float* l2_Wr_r  = (const float*)d_in[18];
  const float* l2_b_r   = (const float*)d_in[19];
  const float* W_out    = (const float*)d_in[20];
  const float* b_out    = (const float*)d_in[21];
  const int* cites_src  = (const int*)d_in[22];
  const int* cites_dst  = (const int*)d_in[23];
  const int* writes_src = (const int*)d_in[24];
  const int* writes_dst = (const int*)d_in[25];
  const int* rev_src    = (const int*)d_in[26];
  const int* rev_dst    = (const int*)d_in[27];
  float* out = (float*)d_out;

  // ---- workspace carve-up (256B aligned) ----
  char* w = (char*)d_ws;
  size_t off = 0;
  auto take = [&](size_t bytes) -> char* {
    char* p = w + off;
    off = (off + bytes + 255) & ~(size_t)255;
    return p;
  };
  float* agg_pc = (float*)take((size_t)NP_ * D_H * 4);
  float* agg_pw = (float*)take((size_t)NP_ * D_H * 4);
  float* agg_ar = (float*)take((size_t)NA_ * D_H * 4);
  float* cnt_pc = (float*)take((size_t)NP_ * 4);
  float* cnt_pw = (float*)take((size_t)NP_ * 4);
  float* cnt_ar = (float*)take((size_t)NA_ * 4);
  char*  end_zero1 = w + off;                       // aggs + counts
  float* h_p  = (float*)take((size_t)NP_ * D_H * 4);
  float* h_a  = (float*)take((size_t)NA_ * D_H * 4);
  bf16*  A_p  = (bf16*)take((size_t)NP_ * (3 * D_H) * 2);
  bf16*  A_a  = (bf16*)take((size_t)NA_ * (2 * D_H) * 2);  // reused as A_out (NP_*256)
  bf16*  Bt_p = (bf16*)take((size_t)D_H * (3 * D_H) * 2);
  bf16*  Bt_a = (bf16*)take((size_t)D_H * (2 * D_H) * 2);
  bf16*  Bt_o = (bf16*)take((size_t)384 * D_H * 2);
  float* bias_p = (float*)take((size_t)D_H * 4);

  const size_t zero1_f4 = (size_t)(end_zero1 - (char*)agg_pc) / 16;       // aggs+cnts
  const size_t zero2_f4 = (size_t)((char*)cnt_pc - (char*)agg_pc) / 16;   // aggs only

  // ============================ LAYER 1 ============================
  k_zero_f32<<<cdiv((long)zero1_f4, 256), 256, 0, stream>>>(agg_pc, zero1_f4);

  k_count<<<cdiv(NE_, 256), 256, 0, stream>>>(cites_dst,  cnt_pc, NE_);
  k_count<<<cdiv(NE_, 256), 256, 0, stream>>>(writes_dst, cnt_pw, NE_);
  k_count<<<cdiv(NE_, 256), 256, 0, stream>>>(rev_dst,    cnt_ar, NE_);

  {
    long n = (long)NE_ * (D_IN / 4);
    k_scatter<D_IN><<<cdiv(n, 256), 256, 0, stream>>>(x_paper,  cites_src,  cites_dst,  agg_pc, NE_);
    k_scatter<D_IN><<<cdiv(n, 256), 256, 0, stream>>>(x_author, writes_src, writes_dst, agg_pw, NE_);
    k_scatter<D_IN><<<cdiv(n, 256), 256, 0, stream>>>(x_paper,  rev_src,    rev_dst,    agg_ar, NE_);
  }

  k_buildA_paper<D_IN><<<cdiv((long)NP_ * D_IN, 256), 256, 0, stream>>>(
      agg_pc, cnt_pc, agg_pw, cnt_pw, x_paper, A_p, NP_);
  k_buildA_author<D_IN><<<cdiv((long)NA_ * D_IN, 256), 256, 0, stream>>>(
      agg_ar, cnt_ar, x_author, A_a, NA_);

  k_buildBt3<D_IN><<<cdiv(D_H * 3 * D_IN, 256), 256, 0, stream>>>(l1_Wl_c, l1_Wl_w, l1_Wr_c, l1_Wr_w, Bt_p);
  k_buildBt2<D_IN><<<cdiv(D_H * 2 * D_IN, 256), 256, 0, stream>>>(l1_Wl_r, l1_Wr_r, Bt_a);
  k_bias_add<<<1, 256, 0, stream>>>(l1_b_c, l1_b_w, bias_p);

  gemm(A_p, Bt_p, bias_p, h_p, NP_, D_H, 3 * D_IN, /*relu=*/1, stream);
  gemm(A_a, Bt_a, l1_b_r, h_a, NA_, D_H, 2 * D_IN, /*relu=*/1, stream);

  // ============================ LAYER 2 ============================
  k_zero_f32<<<cdiv((long)zero2_f4, 256), 256, 0, stream>>>(agg_pc, zero2_f4);

  {
    long n = (long)NE_ * (D_H / 4);
    k_scatter<D_H><<<cdiv(n, 256), 256, 0, stream>>>(h_p, cites_src,  cites_dst,  agg_pc, NE_);
    k_scatter<D_H><<<cdiv(n, 256), 256, 0, stream>>>(h_a, writes_src, writes_dst, agg_pw, NE_);
    k_scatter<D_H><<<cdiv(n, 256), 256, 0, stream>>>(h_p, rev_src,    rev_dst,    agg_ar, NE_);
  }

  k_buildA_paper<D_H><<<cdiv((long)NP_ * D_H, 256), 256, 0, stream>>>(
      agg_pc, cnt_pc, agg_pw, cnt_pw, h_p, A_p, NP_);
  k_buildA_author<D_H><<<cdiv((long)NA_ * D_H, 256), 256, 0, stream>>>(
      agg_ar, cnt_ar, h_a, A_a, NA_);

  k_buildBt3<D_H><<<cdiv(D_H * 3 * D_H, 256), 256, 0, stream>>>(l2_Wl_c, l2_Wl_w, l2_Wr_c, l2_Wr_w, Bt_p);
  k_buildBt2<D_H><<<cdiv(D_H * 2 * D_H, 256), 256, 0, stream>>>(l2_Wl_r, l2_Wr_r, Bt_a);
  k_bias_add<<<1, 256, 0, stream>>>(l2_b_c, l2_b_w, bias_p);

  gemm(A_p, Bt_p, bias_p, h_p, NP_, D_H, 3 * D_H, /*relu=*/1, stream);   // overwrites h_p (already consumed)
  gemm(A_a, Bt_a, l2_b_r, h_a, NA_, D_H, 2 * D_H, /*relu=*/1, stream);

  // ============================ OUTPUT ============================
  bf16* A_out = A_a;  // NA_*2*D_H bf16 == NP_*D_H bf16 — safe reuse
  k_cvt_bf16<<<cdiv((long)NP_ * D_H, 256), 256, 0, stream>>>(h_p, A_out, (long)NP_ * D_H);
  k_buildBtOut<<<cdiv(384 * D_H, 256), 256, 0, stream>>>(W_out, Bt_o);

  gemm(A_out, Bt_o, b_out, out, NP_, D_OUT, D_H, /*relu=*/0, stream);
}
